// SingleScaleSplitModel_35588099014900
// MI455X (gfx1250) — compile-verified, matrix operand order checked by reading, not run on hardware
//
#include <hip/hip_runtime.h>

// MI455X / gfx1250, wave32. WMMA 16x16x32 f16->f32 + async global->LDS path.

typedef __attribute__((ext_vector_type(16))) _Float16 v16h;
typedef __attribute__((ext_vector_type(8)))  float    v8f;

#define B_    256
#define L_    4096
#define V_    64
#define H_    64
#define HALF_ 32

__device__ __forceinline__ void lds_fence() {
  asm volatile("s_wait_dscnt 0" ::: "memory");
}

// A-operand (or B-from-transposed-storage) loader: f16 row-major LDS, row = rowoff+ln,
// K-pattern per ISA 16-bit A 16x32 layout: reg r packs K = kb, kb+1 with
// kb = (r<4 ? 2r : 16+2(r-4)) + 8*(lane>=16) + koff.
__device__ __forceinline__ v16h load_a16(const _Float16* p, int ld, int rowoff, int koff,
                                         int ln, int hf) {
  v16h a;
#pragma unroll
  for (int r = 0; r < 8; ++r) {
    int kb = (r < 4 ? 2 * r : 16 + 2 * (r - 4)) + hf * 8 + koff;
    const _Float16* q = p + (rowoff + ln) * ld + kb;
    a[2 * r]     = q[0];
    a[2 * r + 1] = q[1];
  }
  return a;
}

// Loader for operands whose K runs down the rows of the LDS buffer:
// element [kb][coloff+ln] (used for DW^T and K as B in the update GEMM).
__device__ __forceinline__ v16h load_strided16(const _Float16* p, int ld, int coloff,
                                               int ln, int hf) {
  v16h a;
#pragma unroll
  for (int r = 0; r < 8; ++r) {
    int kb = (r < 4 ? 2 * r : 16 + 2 * (r - 4)) + hf * 8;
    a[2 * r]     = p[kb * ld + coloff + ln];
    a[2 * r + 1] = p[(kb + 1) * ld + coloff + ln];
  }
  return a;
}

__device__ __forceinline__ v8f wmma16(v16h a, v16h b, v8f c) {
  return __builtin_amdgcn_wmma_f32_16x16x32_f16(false, a, false, b, (short)0, c, false, false);
}

// ---------------------------------------------------------------------------
// Kernel 1: encoder. 4 waves/block, each wave owns a 16-token tile.
// e=embed[seq]; z1=relu(e@W1+b1); f=z1@W2+b2; h=LN(e+f)*g+b; k=h@W{sem,epi}+b.
// ---------------------------------------------------------------------------
__global__ __launch_bounds__(128) void encoder_kernel(
    const int* __restrict__ seq, const float* __restrict__ embed,
    const float* __restrict__ W1, const float* __restrict__ b1,
    const float* __restrict__ W2, const float* __restrict__ b2,
    const float* __restrict__ gamma, const float* __restrict__ beta,
    const float* __restrict__ Wsem, const float* __restrict__ bsem,
    const float* __restrict__ Wepi, const float* __restrict__ bepi,
    float* __restrict__ kall_s, float* __restrict__ kall_e) {
  __shared__ _Float16 sW1T[128 * 64];   // W1T[n][k] = W1[k][n]
  __shared__ _Float16 sW2T[64 * 128];   // W2T[n][k] = W2[k][n]
  __shared__ _Float16 sWsT[32 * 64];
  __shared__ _Float16 sWeT[32 * 64];
  __shared__ float sb1[128], sb2[64], sg[64], sbt[64], sbs[32], sbe[32];
  __shared__ _Float16 sbuf[4][16 * 128];  // per-wave scratch (e -> z1 -> h)
  __shared__ int stok[4][16];

  const int tid = threadIdx.x;
  for (int i = tid; i < 64 * 128; i += 128) { int k = i >> 7, n = i & 127; sW1T[n * 64 + k] = (_Float16)W1[i]; }
  for (int i = tid; i < 128 * 64; i += 128) { int k = i >> 6, n = i & 63;  sW2T[n * 128 + k] = (_Float16)W2[i]; }
  for (int i = tid; i < 64 * 32;  i += 128) { int k = i >> 5, n = i & 31;
    sWsT[n * 64 + k] = (_Float16)Wsem[i]; sWeT[n * 64 + k] = (_Float16)Wepi[i]; }
  if (tid < 128) sb1[tid] = b1[tid];
  if (tid < 64) { sb2[tid] = b2[tid]; sg[tid] = gamma[tid]; sbt[tid] = beta[tid]; }
  if (tid < 32) { sbs[tid] = bsem[tid]; sbe[tid] = bepi[tid]; }
  __syncthreads();

  const int wave = tid >> 5, lane = tid & 31;
  const int hf = lane >> 4, ln = lane & 15;
  const long t0 = ((long)blockIdx.x * 4 + wave) * 16;  // flat token start (b*L + l)
  _Float16* buf = sbuf[wave];
  int* tok = stok[wave];
  if (lane < 16) tok[lane] = (int)seq[t0 + lane];
  lds_fence();

  // e tile -> LDS f16 [16][64]
  for (int r = 0; r < 16; ++r) {
    int t = tok[r];
    buf[r * 64 + lane]      = (_Float16)embed[t * 64 + lane];
    buf[r * 64 + lane + 32] = (_Float16)embed[t * 64 + lane + 32];
  }
  lds_fence();

  // z1 = relu(e @ W1 + b1)  [16,128]
  v8f z[8];
#pragma unroll
  for (int tj = 0; tj < 8; ++tj) z[tj] = (v8f){};
#pragma unroll
  for (int ks = 0; ks < 2; ++ks) {
    v16h a = load_a16(buf, 64, 0, ks * 32, ln, hf);
#pragma unroll
    for (int tj = 0; tj < 8; ++tj)
      z[tj] = wmma16(a, load_a16(sW1T, 64, 16 * tj, ks * 32, ln, hf), z[tj]);
  }
  lds_fence();
#pragma unroll
  for (int tj = 0; tj < 8; ++tj)
#pragma unroll
    for (int r = 0; r < 8; ++r) {
      int row = r + 8 * hf, col = 16 * tj + ln;
      buf[row * 128 + col] = (_Float16)fmaxf(z[tj][r] + sb1[col], 0.0f);
    }
  lds_fence();

  // f = z1 @ W2 + b2 ; y = e + f ; h = LN(y)
  v8f f[4];
#pragma unroll
  for (int tj = 0; tj < 4; ++tj) f[tj] = (v8f){};
#pragma unroll
  for (int ks = 0; ks < 4; ++ks) {
    v16h a = load_a16(buf, 128, 0, ks * 32, ln, hf);
#pragma unroll
    for (int tj = 0; tj < 4; ++tj)
      f[tj] = wmma16(a, load_a16(sW2T, 128, 16 * tj, ks * 32, ln, hf), f[tj]);
  }
  float y[4][8], rs1[8], rs2[8];
#pragma unroll
  for (int r = 0; r < 8; ++r) { rs1[r] = 0.f; rs2[r] = 0.f; }
#pragma unroll
  for (int tj = 0; tj < 4; ++tj)
#pragma unroll
    for (int r = 0; r < 8; ++r) {
      int row = r + 8 * hf, col = 16 * tj + ln;
      float e32 = embed[tok[row] * 64 + col];          // f32 residual path
      float v = f[tj][r] + sb2[col] + e32;
      y[tj][r] = v; rs1[r] += v; rs2[r] += v * v;
    }
#pragma unroll
  for (int r = 0; r < 8; ++r)
    for (int m = 8; m >= 1; m >>= 1) {
      rs1[r] += __shfl_xor(rs1[r], m, 32);
      rs2[r] += __shfl_xor(rs2[r], m, 32);
    }
  lds_fence();
#pragma unroll
  for (int tj = 0; tj < 4; ++tj)
#pragma unroll
    for (int r = 0; r < 8; ++r) {
      int row = r + 8 * hf, col = 16 * tj + ln;
      float mu = rs1[r] * (1.0f / 64.0f);
      float var = rs2[r] * (1.0f / 64.0f) - mu * mu;
      float h = (y[tj][r] - mu) * rsqrtf(var + 1e-5f) * sg[col] + sbt[col];
      buf[row * 64 + col] = (_Float16)h;
    }
  lds_fence();

  // kall_{s,e} = h @ W{sem,epi} + b   [16,32] each, stored f32
#pragma unroll
  for (int tj = 0; tj < 2; ++tj) {
    v8f cs = (v8f){}, ce = (v8f){};
#pragma unroll
    for (int ks = 0; ks < 2; ++ks) {
      v16h a = load_a16(buf, 64, 0, ks * 32, ln, hf);
      cs = wmma16(a, load_a16(sWsT, 64, 16 * tj, ks * 32, ln, hf), cs);
      ce = wmma16(a, load_a16(sWeT, 64, 16 * tj, ks * 32, ln, hf), ce);
    }
#pragma unroll
    for (int r = 0; r < 8; ++r) {
      int row = r + 8 * hf, col = 16 * tj + ln;
      kall_s[(t0 + row) * HALF_ + col] = cs[r] + sbs[col];
      kall_e[(t0 + row) * HALF_ + col] = ce[r] + sbe[col];
    }
  }
}

// ---------------------------------------------------------------------------
// Kernel 2: chunked delta-rule scan. One block per batch, wave0=semantic,
// wave1=episodic. State M (32x32) lives in C-layout WMMA accumulators.
// Key chunks are fetched with double-buffered GLOBAL_LOAD_ASYNC_TO_LDS_B128
// (ASYNCcnt), overlapping HBM/L2 latency with the WMMA + solve of the
// current chunk.
// ---------------------------------------------------------------------------
__global__ __launch_bounds__(64) void scan_kernel(
    const float* __restrict__ kall_s, const float* __restrict__ kall_e,
    const float* __restrict__ Wout, const float* __restrict__ bout,
    float* __restrict__ out) {
  __shared__ float    sK32[2][2][16 * 32];  // [mem][ping-pong] f32 key chunk
  __shared__ _Float16 sK16[2][32 * 32];     // rows 16..31 stay zero (K-dim pad)
  __shared__ _Float16 sM16[2][32 * 32];
  __shared__ _Float16 sDW [2][32 * 32];     // rows 16..31 stay zero
  __shared__ float    sG  [2][16 * 16];
  __shared__ float    sP  [2][16 * 32];     // P[t][row]
  __shared__ float    sCat[64];

  const int mem  = threadIdx.x >> 5;  // 0=semantic, 1=episodic
  const int lane = threadIdx.x & 31;
  const int hf = lane >> 4, ln = lane & 15;
  const int b = blockIdx.x;
  const float* kall = mem ? kall_e : kall_s;
  _Float16* K16 = sK16[mem];
  _Float16* M16 = sM16[mem];
  _Float16* DW  = sDW[mem];
  float*    G   = sG[mem];
  float*    P   = sP[mem];

  for (int t = 16; t < 32; ++t) { K16[t * 32 + lane] = (_Float16)0.f; DW[t * 32 + lane] = (_Float16)0.f; }

  v8f M[4];  // 2x2 tiles of the 32x32 state, C-layout
#pragma unroll
  for (int i = 0; i < 4; ++i) M[i] = (v8f){};

  const long base = (long)b * L_;
  const float invL = 1.0f / (float)L_;
  v8f zero = (v8f){};

  // Async fetch of one 16x32 f32 key chunk (2048B) into LDS buffer pb.
  // Per-lane 16B x 4 instructions; GVS addressing (saddr = chunk base).
  const unsigned voff = (unsigned)(lane * 16);
  auto issue_chunk = [&](int c0, int pb) {
    unsigned ldsv = (unsigned)(size_t)(void*)sK32[mem][pb] + voff;
    const float* gp = &kall[(base + c0) * HALF_];
    asm volatile("global_load_async_to_lds_b128 %0, %1, %2"             :: "v"(ldsv), "v"(voff), "s"(gp) : "memory");
    asm volatile("global_load_async_to_lds_b128 %0, %1, %2 offset:512"  :: "v"(ldsv), "v"(voff), "s"(gp) : "memory");
    asm volatile("global_load_async_to_lds_b128 %0, %1, %2 offset:1024" :: "v"(ldsv), "v"(voff), "s"(gp) : "memory");
    asm volatile("global_load_async_to_lds_b128 %0, %1, %2 offset:1536" :: "v"(ldsv), "v"(voff), "s"(gp) : "memory");
  };

  issue_chunk(0, 0);
  int pb = 0;

  for (int c0 = 0; c0 < L_ - 1; c0 += 16) {
    const int cnt = ((L_ - 1) - c0 < 16) ? (L_ - 1) - c0 : 16;
    // prefetch two chunks ahead into L2 (beyond the double buffer)
    if (c0 + 32 < L_ - 1)
      __builtin_prefetch(&kall[(base + c0 + 32) * HALF_], 0, 0);
    // overlap: issue next chunk, then wait for the current one (in-order
    // completion: <=1 outstanding means the current chunk has landed).
    if (c0 + 16 < L_ - 1) {
      issue_chunk(c0 + 16, pb ^ 1);
      asm volatile("s_wait_asynccnt 1" ::: "memory");
    } else {
      asm volatile("s_wait_asynccnt 0" ::: "memory");
    }
    float* K32 = sK32[mem][pb];

    // f16 mirror of the chunk for WMMA operands. Rows >= cnt may hold the
    // query key (in-bounds read) -- harmless: dw is forced to 0 there, so
    // those rows never touch the state update.
#pragma unroll 4
    for (int t = 0; t < 16; ++t)
      K16[t * 32 + lane] = (_Float16)K32[t * 32 + lane];

    // dump state as f16 for the A-operand of P = M K^T
#pragma unroll
    for (int i = 0; i < 4; ++i) {
      int ti = i >> 1, tj = i & 1;
#pragma unroll
      for (int r = 0; r < 8; ++r)
        M16[(16 * ti + r + 8 * hf) * 32 + 16 * tj + ln] = (_Float16)M[i][r];
    }
    lds_fence();

    // G = K K^T (one WMMA: A row m = key m, B col n = key n -> same load)
    v16h kk = load_a16(K16, 32, 0, 0, ln, hf);
    v8f g = wmma16(kk, kk, zero);
#pragma unroll
    for (int r = 0; r < 8; ++r) G[(r + 8 * hf) * 16 + ln] = g[r];

    // P = M K^T  [32,16]
#pragma unroll
    for (int ti = 0; ti < 2; ++ti) {
      v8f p = wmma16(load_a16(M16, 32, 16 * ti, 0, ln, hf), kk, zero);
#pragma unroll
      for (int r = 0; r < 8; ++r) P[ln * 32 + 16 * ti + r + 8 * hf] = p[r];
    }
    lds_fence();

    // triangular solve: dw_t = c_t * (k_t - (P[:,t] + sum_{s<t} G[s,t] dw_s)/(G[t,t]+eps))
    float acc[16];
#pragma unroll
    for (int t = 0; t < 16; ++t) acc[t] = P[t * 32 + lane];
#pragma unroll
    for (int t = 0; t < 16; ++t) {
      float nt = G[t * 16 + t] + 1e-6f;
      float dv = K32[t * 32 + lane] - acc[t] / nt;
      float cf = mem ? (float)(c0 + t + 1) * invL : 1.0f;
      float dw = (t < cnt) ? cf * dv : 0.0f;
      DW[t * 32 + lane] = (_Float16)dw;
#pragma unroll
      for (int u = t + 1; u < 16; ++u) acc[u] += G[t * 16 + u] * dw;
    }
    lds_fence();

    // M += DW^T K  (inner dim padded 16->32 with zero rows)
#pragma unroll
    for (int i = 0; i < 4; ++i) {
      int ti = i >> 1, tj = i & 1;
      v16h au = load_strided16(DW,  32, 16 * ti, ln, hf);
      v16h bu = load_strided16(K16, 32, 16 * tj, ln, hf);
      M[i] = wmma16(au, bu, M[i]);
    }
    lds_fence();
    pb ^= 1;
  }

  // query with the last key (column 0 of a zero-padded chunk)
  for (int t = 0; t < 16; ++t)
    K16[t * 32 + lane] = (t == 0) ? (_Float16)kall[(base + (L_ - 1)) * HALF_ + lane]
                                  : (_Float16)0.f;
#pragma unroll
  for (int i = 0; i < 4; ++i) {
    int ti = i >> 1, tj = i & 1;
#pragma unroll
    for (int r = 0; r < 8; ++r)
      M16[(16 * ti + r + 8 * hf) * 32 + 16 * tj + ln] = (_Float16)M[i][r];
  }
  lds_fence();
  {
    v16h kk = load_a16(K16, 32, 0, 0, ln, hf);
#pragma unroll
    for (int ti = 0; ti < 2; ++ti) {
      v8f p = wmma16(load_a16(M16, 32, 16 * ti, 0, ln, hf), kk, zero);
#pragma unroll
      for (int r = 0; r < 8; ++r) P[ln * 32 + 16 * ti + r + 8 * hf] = p[r];
    }
  }
  lds_fence();
  sCat[mem * 32 + lane] = P[lane];  // column t=0
  __syncthreads();

  // out[b] = concat(cs,ce) @ Wout + bout   (64x64 GEMV)
  {
    int v = threadIdx.x;  // 64 threads == 64 outputs
    float acc = bout[v];
    for (int ff = 0; ff < 2 * HALF_; ++ff) acc += sCat[ff] * Wout[ff * V_ + v];
    out[b * V_ + v] = acc;
  }
}

// ---------------------------------------------------------------------------
extern "C" void kernel_launch(void* const* d_in, const int* in_sizes, int n_in,
                              void* d_out, int out_size, void* d_ws, size_t ws_size,
                              hipStream_t stream) {
  const int*   seq   = (const int*)d_in[0];
  const float* embed = (const float*)d_in[1];
  const float* W1    = (const float*)d_in[2];
  const float* b1    = (const float*)d_in[3];
  const float* W2    = (const float*)d_in[4];
  const float* b2    = (const float*)d_in[5];
  const float* gamma = (const float*)d_in[6];
  const float* beta  = (const float*)d_in[7];
  const float* Wsem  = (const float*)d_in[8];
  const float* bsem  = (const float*)d_in[9];
  const float* Wepi  = (const float*)d_in[10];
  const float* bepi  = (const float*)d_in[11];
  const float* Wout  = (const float*)d_in[12];
  const float* bout  = (const float*)d_in[13];

  float* kall_s = (float*)d_ws;
  float* kall_e = kall_s + (size_t)B_ * L_ * HALF_;

  const int enc_blocks = (B_ * L_) / 64;  // 4 waves/block * 16 tokens/wave
  encoder_kernel<<<enc_blocks, 128, 0, stream>>>(seq, embed, W1, b1, W2, b2, gamma, beta,
                                                 Wsem, bsem, Wepi, bepi, kall_s, kall_e);
  scan_kernel<<<B_, 64, 0, stream>>>(kall_s, kall_e, Wout, bout, (float*)d_out);
}
